// GptOssExperts_57354993271421
// MI455X (gfx1250) — matrix operand order repaired and checked by reading, not compiled, not run on hardware
//
#include <hip/hip_runtime.h>
#include <hip/hip_bf16.h>

typedef __attribute__((ext_vector_type(16))) __bf16 bf16x16;
typedef __attribute__((ext_vector_type(8)))  float  floatx8;

namespace {
constexpr int S = 1024, H = 2880, I = 2880, E = 16, TOPK = 4;
constexpr int T = S * TOPK;                 // 4096 routed (token,expert) entries
constexpr float LIMIT = 7.0f, ALPHA = 1.702f;

constexpr int MT = 128;                     // M tile (tokens per WG)
constexpr int KS = 64;                      // K step
constexpr int LDA = 72;                     // LDS strides (halves), padded
constexpr int LDB = 72;

// ---- workspace layout (bytes) ----
constexpr size_t OFF_COUNTS = 0;                           // E ints
constexpr size_t OFF_FILL   = 64;                          // E ints
constexpr size_t OFF_OFFS   = 128;                         // E+1 ints
constexpr size_t OFF_TOK    = 256;                         // T ints
constexpr size_t OFF_WGT    = OFF_TOK + (size_t)T * 4;     // T floats
constexpr size_t OFF_XBF    = 33024;                       // S*H bf16
constexpr size_t OFF_HBUF   = OFF_XBF + (size_t)S * H * 2; // T*I bf16
} // namespace

// ---------------- helpers ----------------
__device__ __forceinline__ unsigned short f32_to_bf16(float f) {
  unsigned int u = __float_as_uint(f);
  u += 0x7FFFu + ((u >> 16) & 1u);           // round-to-nearest-even
  return (unsigned short)(u >> 16);
}
__device__ __forceinline__ unsigned int pack2_bf16(float lo, float hi) {
  return (unsigned int)f32_to_bf16(lo) | ((unsigned int)f32_to_bf16(hi) << 16);
}
__device__ __forceinline__ unsigned lds_off(const void* p) {
  return (unsigned)(uintptr_t)p;             // low 32 bits of flat LDS ptr = LDS offset
}
// CDNA5 async copy: global -> LDS, 16B per lane, tracked by ASYNCcnt
__device__ __forceinline__ void async_ld_b128(unsigned lds, const unsigned short* g) {
  asm volatile("global_load_async_to_lds_b128 %0, %1, off" :: "v"(lds), "v"(g) : "memory");
}
__device__ __forceinline__ void wait_async0() {
  asm volatile("s_wait_asynccnt 0x0" ::: "memory");
}
// B fragment: 16 contiguous halves
__device__ __forceinline__ bf16x16 ld_frag_b(const unsigned short* p) {
  bf16x16 v; uint4* q = reinterpret_cast<uint4*>(&v);
  q[0] = *reinterpret_cast<const uint4*>(p);
  q[1] = *reinterpret_cast<const uint4*>(p + 8);
  return v;
}
// A fragment: 16 contiguous halves (kh folded into p)
__device__ __forceinline__ bf16x16 ld_frag_a(const unsigned short* p) {
  bf16x16 v; uint4* q = reinterpret_cast<uint4*>(&v);
  q[0] = *reinterpret_cast<const uint4*>(p);
  q[1] = *reinterpret_cast<const uint4*>(p + 8);
  return v;
}
__device__ __forceinline__ floatx8 wmma_bf16(bf16x16 a, bf16x16 b, floatx8 c) {
  return __builtin_amdgcn_wmma_f32_16x16x32_bf16(false, a, false, b, (short)0, c, false, false);
}

// ---------------- routing ----------------
__global__ void k_route_init(int* counts, int* fill) {
  if (threadIdx.x < E) { counts[threadIdx.x] = 0; fill[threadIdx.x] = 0; }
}
__global__ void k_route_count(const int* __restrict__ ridx, int* counts) {
  int idx = blockIdx.x * blockDim.x + threadIdx.x;
  if (idx < T) atomicAdd(&counts[ridx[idx]], 1);
}
__global__ void k_route_scan(const int* __restrict__ counts, int* offs) {
  if (threadIdx.x == 0) {
    int acc = 0;
    for (int e = 0; e < E; ++e) { offs[e] = acc; acc += counts[e]; }
    offs[E] = acc;
  }
}
__global__ void k_route_fill(const int* __restrict__ ridx,
                             const float* __restrict__ rwgt,
                             const int* __restrict__ offs, int* fill,
                             int* tok, float* wgt) {
  int idx = blockIdx.x * blockDim.x + threadIdx.x;
  if (idx < T) {
    int s = idx >> 2;
    int e = ridx[idx];
    int p = offs[e] + atomicAdd(&fill[e], 1);
    tok[p] = s;
    wgt[p] = rwgt[idx];
  }
}

// ---------------- misc elementwise ----------------
__global__ void k_zero_out(float* out) {
  int idx = blockIdx.x * blockDim.x + threadIdx.x;
  float4 z = {0.f, 0.f, 0.f, 0.f};
  reinterpret_cast<float4*>(out)[idx] = z;
}
__global__ void k_cvt_x(const float* __restrict__ x, unsigned short* __restrict__ xbf) {
  int idx = blockIdx.x * blockDim.x + threadIdx.x;
  float4 f = reinterpret_cast<const float4*>(x)[idx];
  uint2 p;
  p.x = pack2_bf16(f.x, f.y);
  p.y = pack2_bf16(f.z, f.w);
  reinterpret_cast<uint2*>(xbf)[idx] = p;
}

// ---------------- GEMM1: gate_up + activation -> h (bf16) ----------------
// grid: (I/64, T/MT, E)  block 256 (8 waves)
__global__ __launch_bounds__(256)
void k_gemm1(const unsigned short* __restrict__ xbf,
             const float* __restrict__ W1, const float* __restrict__ b1,
             const int* __restrict__ offs, const int* __restrict__ tok,
             unsigned short* __restrict__ hbuf) {
  __shared__ __align__(16) unsigned short As[2][MT * LDA];
  __shared__ __align__(16) unsigned short Bs[128 * LDB]; // rows 0..63 gate, 64..127 up

  const int e  = blockIdx.z;
  const int mt = blockIdx.y;
  const int nt = blockIdx.x;
  const int base = offs[e];
  const int ne   = offs[e + 1] - base;
  if (mt * MT >= ne) return;
  const int rcount = min(MT, ne - mt * MT);

  const int t    = threadIdx.x;
  const int lane = t & 31;
  const int wave = t >> 5;
  const int ni    = wave & 3;          // N subtile 0..3
  const int waveM = (wave >> 2) * 64;  // M half 0 / 64

  // --- A staging (async): thread covers halves [(t&1)*32, +32) of row t>>1 ---
  const int ar  = t >> 1;
  const int ak  = (t & 1) * 32;
  const int arr = ar < rcount ? ar : (rcount - 1);
  const unsigned short* agbase =
      xbf + (long)tok[base + mt * MT + arr] * H + ak;
  const unsigned myA0 = lds_off(&As[0][ar * LDA + ak]);
  const unsigned myA1 = lds_off(&As[1][ar * LDA + ak]);

  // --- B staging (regs + cvt): row nall = t>>1 (0..127), K chunk (t&1)*32 floats ---
  const int nall = t >> 1;
  const int bn_global = (nall < 64) ? (nt * 64 + nall) : (I + nt * 64 + (nall - 64));
  const float* bptr = W1 + ((long)e * (2 * I) + bn_global) * H + (t & 1) * 32;
  unsigned short* bdst = &Bs[nall * LDB + (t & 1) * 32];

  floatx8 cg[4] = {{}, {}, {}, {}};
  floatx8 cu[4] = {{}, {}, {}, {}};
  float4 breg[8];

  auto issueA = [&](int k, int buf) {
    unsigned d = buf ? myA1 : myA0;
    const unsigned short* g = agbase + k;
#pragma unroll
    for (int c = 0; c < 4; ++c) async_ld_b128(d + c * 16, g + c * 8);
  };
  auto loadB = [&](int k) {
    const float4* p = reinterpret_cast<const float4*>(bptr + k);
#pragma unroll
    for (int q = 0; q < 8; ++q) breg[q] = p[q];
    __builtin_prefetch(bptr + ((k + 2 * KS) % H), 0, 1);
  };
  auto storeB = [&]() {
#pragma unroll
    for (int q = 0; q < 4; ++q) {
      uint4 w;
      w.x = pack2_bf16(breg[2 * q].x, breg[2 * q].y);
      w.y = pack2_bf16(breg[2 * q].z, breg[2 * q].w);
      w.z = pack2_bf16(breg[2 * q + 1].x, breg[2 * q + 1].y);
      w.w = pack2_bf16(breg[2 * q + 1].z, breg[2 * q + 1].w);
      *reinterpret_cast<uint4*>(bdst + q * 8) = w;
    }
  };

  constexpr int NIT = H / KS;  // 45
  issueA(0, 0);
  loadB(0);

  for (int it = 0; it < NIT; ++it) {
    const int cur = it & 1;
    const int kn  = ((it + 1) < NIT) ? (it + 1) * KS : 0;  // wrap: harmless, never consumed
    storeB();
    wait_async0();              // A[cur] resident
    __syncthreads();
    issueA(kn, cur ^ 1);        // overlap next A copy with compute
    loadB(kn);                  // overlap next B load with compute
    const int kh = (lane >> 4) * 16;
#pragma unroll
    for (int kk = 0; kk < KS; kk += 32) {
      const unsigned short* bb = &Bs[(ni * 16 + (lane & 15)) * LDB + kk + (lane >> 4) * 16];
      bf16x16 bg = ld_frag_b(bb);
      bf16x16 bu = ld_frag_b(bb + 64 * LDB);
#pragma unroll
      for (int m = 0; m < 4; ++m) {
        bf16x16 a = ld_frag_a(&As[cur][(waveM + m * 16 + (lane & 15)) * LDA + kk + kh]);
        cg[m] = wmma_bf16(a, bg, cg[m]);
        cu[m] = wmma_bf16(a, bu, cu[m]);
      }
    }
    __syncthreads();
  }

  // ---- epilogue: bias + clamp + GLU, store bf16 h ----
  const int col  = lane & 15;
  const int gcol = nt * 64 + ni * 16 + col;
  const float big = b1[(long)e * (2 * I) + gcol];
  const float biu = b1[(long)e * (2 * I) + I + gcol];
  unsigned short* hbase = hbuf + (long)(base + mt * MT) * I;
  const int rh = (lane >> 4) * 8;
#pragma unroll
  for (int m = 0; m < 4; ++m) {
#pragma unroll
    for (int r = 0; r < 8; ++r) {
      int row_l = waveM + m * 16 + rh + r;
      if (row_l < rcount) {
        float g = fminf(cg[m][r] + big, LIMIT);
        float u = fminf(fmaxf(cu[m][r] + biu, -LIMIT), LIMIT);
        float glu = g * __builtin_amdgcn_rcpf(1.0f + __expf(-ALPHA * g));
        hbase[(long)row_l * I + gcol] = f32_to_bf16((u + 1.0f) * glu);
      }
    }
  }
}

// ---------------- GEMM2: down proj + weighted scatter ----------------
// grid: (H/64, T/MT, E)  block 256
__global__ __launch_bounds__(256)
void k_gemm2(const unsigned short* __restrict__ hbuf,
             const float* __restrict__ W2, const float* __restrict__ b2,
             const int* __restrict__ offs, const int* __restrict__ tok,
             const float* __restrict__ wgt, float* __restrict__ out) {
  __shared__ __align__(16) unsigned short As[2][MT * LDA];
  __shared__ __align__(16) unsigned short Bs[64 * LDB];

  const int e  = blockIdx.z;
  const int mt = blockIdx.y;
  const int nt = blockIdx.x;
  const int base = offs[e];
  const int ne   = offs[e + 1] - base;
  if (mt * MT >= ne) return;
  const int rcount = min(MT, ne - mt * MT);

  const int t    = threadIdx.x;
  const int lane = t & 31;
  const int wave = t >> 5;
  const int ni    = wave & 3;
  const int waveM = (wave >> 2) * 64;

  const int ar  = t >> 1;
  const int ak  = (t & 1) * 32;
  const int arr = ar < rcount ? ar : (rcount - 1);
  const unsigned short* agbase = hbuf + (long)(base + mt * MT + arr) * I + ak;
  const unsigned myA0 = lds_off(&As[0][ar * LDA + ak]);
  const unsigned myA1 = lds_off(&As[1][ar * LDA + ak]);

  // B staging: row n = t>>2 (0..63), K chunk (t&3)*16 floats
  const int bn = t >> 2;
  const float* bptr = W2 + ((long)e * H + nt * 64 + bn) * I + (t & 3) * 16;
  unsigned short* bdst = &Bs[bn * LDB + (t & 3) * 16];

  floatx8 c[4] = {{}, {}, {}, {}};
  float4 breg[4];

  auto issueA = [&](int k, int buf) {
    unsigned d = buf ? myA1 : myA0;
    const unsigned short* g = agbase + k;
#pragma unroll
    for (int cc = 0; cc < 4; ++cc) async_ld_b128(d + cc * 16, g + cc * 8);
  };
  auto loadB = [&](int k) {
    const float4* p = reinterpret_cast<const float4*>(bptr + k);
#pragma unroll
    for (int q = 0; q < 4; ++q) breg[q] = p[q];
    __builtin_prefetch(bptr + ((k + 2 * KS) % I), 0, 1);
  };
  auto storeB = [&]() {
#pragma unroll
    for (int q = 0; q < 2; ++q) {
      uint4 w;
      w.x = pack2_bf16(breg[2 * q].x, breg[2 * q].y);
      w.y = pack2_bf16(breg[2 * q].z, breg[2 * q].w);
      w.z = pack2_bf16(breg[2 * q + 1].x, breg[2 * q + 1].y);
      w.w = pack2_bf16(breg[2 * q + 1].z, breg[2 * q + 1].w);
      *reinterpret_cast<uint4*>(bdst + q * 8) = w;
    }
  };

  constexpr int NIT = I / KS;  // 45
  issueA(0, 0);
  loadB(0);

  for (int it = 0; it < NIT; ++it) {
    const int cur = it & 1;
    const int kn  = ((it + 1) < NIT) ? (it + 1) * KS : 0;
    storeB();
    wait_async0();
    __syncthreads();
    issueA(kn, cur ^ 1);
    loadB(kn);
    const int kh = (lane >> 4) * 16;
#pragma unroll
    for (int kk = 0; kk < KS; kk += 32) {
      bf16x16 b = ld_frag_b(&Bs[(ni * 16 + (lane & 15)) * LDB + kk + (lane >> 4) * 16]);
#pragma unroll
      for (int m = 0; m < 4; ++m) {
        bf16x16 a = ld_frag_a(&As[cur][(waveM + m * 16 + (lane & 15)) * LDA + kk + kh]);
        c[m] = wmma_bf16(a, b, c[m]);
      }
    }
    __syncthreads();
  }

  const int col  = lane & 15;
  const int gcol = nt * 64 + ni * 16 + col;
  const float bias = b2[(long)e * H + gcol];
  const int rh = (lane >> 4) * 8;
#pragma unroll
  for (int m = 0; m < 4; ++m) {
#pragma unroll
    for (int r = 0; r < 8; ++r) {
      int row_l = waveM + m * 16 + rh + r;
      if (row_l < rcount) {
        int entry = base + mt * MT + row_l;
        float y = (c[m][r] + bias) * wgt[entry];
        atomicAdd(&out[(long)tok[entry] * H + gcol], y);
      }
    }
  }
}

// ---------------- launch ----------------
extern "C" void kernel_launch(void* const* d_in, const int* in_sizes, int n_in,
                              void* d_out, int out_size, void* d_ws, size_t ws_size,
                              hipStream_t stream) {
  const float* x    = (const float*)d_in[0];  // [1,S,H]
  const int*   ridx = (const int*)d_in[1];    // [S,K]
  const float* rwgt = (const float*)d_in[2];  // [S,K]
  const float* W1   = (const float*)d_in[3];  // [E,2I,H]
  const float* b1   = (const float*)d_in[4];  // [E,2I]
  const float* W2   = (const float*)d_in[5];  // [E,H,I]
  const float* b2   = (const float*)d_in[6];  // [E,H]
  float* out = (float*)d_out;

  char* ws = (char*)d_ws;
  int*   counts = (int*)(ws + OFF_COUNTS);
  int*   fill   = (int*)(ws + OFF_FILL);
  int*   offs   = (int*)(ws + OFF_OFFS);
  int*   tokl   = (int*)(ws + OFF_TOK);
  float* wgtl   = (float*)(ws + OFF_WGT);
  unsigned short* xbf  = (unsigned short*)(ws + OFF_XBF);
  unsigned short* hbuf = (unsigned short*)(ws + OFF_HBUF);

  k_route_init<<<1, 32, 0, stream>>>(counts, fill);
  k_route_count<<<T / 256, 256, 0, stream>>>(ridx, counts);
  k_route_scan<<<1, 32, 0, stream>>>(counts, offs);
  k_route_fill<<<T / 256, 256, 0, stream>>>(ridx, rwgt, offs, fill, tokl, wgtl);

  k_zero_out<<<(S * H / 4) / 256, 256, 0, stream>>>(out);
  k_cvt_x<<<(S * H / 4) / 256, 256, 0, stream>>>(x, xbf);

  dim3 g1(I / 64, T / MT, E);
  k_gemm1<<<g1, 256, 0, stream>>>(xbf, W1, b1, offs, tokl, hbuf);
  dim3 g2(H / 64, T / MT, E);
  k_gemm2<<<g2, 256, 0, stream>>>(hbuf, W2, b2, offs, tokl, wgtl, out);
}
